// MoELayer_85718957294255
// MI455X (gfx1250) — matrix operand order repaired and checked by reading, not compiled, not run on hardware
//
#include <hip/hip_runtime.h>
#include <hip/hip_bf16.h>
#include <math.h>

#define T_TOK   8192
#define D_MODEL 2048
#define N_EXP   8
#define LOSS_SCALE 3e-06f
#define PERM_CAP (T_TOK + 32 * N_EXP)     // expert groups padded to 32 rows
#define NT16 528                           // max 16-row tiles (fallback path)
#define NT32 264                           // max 32-row tiles (split path)
#define BM 32                              // block M tile (split path)
#define BN 256                             // block N tile (split path)
#define LDB 40                             // bf16 per LDS row: 32 data + 8 pad = 80B

typedef __attribute__((ext_vector_type(16))) __bf16 v16bf;
typedef __attribute__((ext_vector_type(8)))  float  v8f;
typedef __attribute__((ext_vector_type(4)))  __bf16 v4bf;

// ---------------------------------------------------------------------------
// Kernel 1: router. One wave per token, fp32 for argmax/softmax fidelity.
// ---------------------------------------------------------------------------
__global__ __launch_bounds__(256) void router_kernel(
    const float* __restrict__ x, const float* __restrict__ Wr,
    const float* __restrict__ br, int* __restrict__ idx,
    float* __restrict__ topp) {
  int token = blockIdx.x * 8 + (threadIdx.x >> 5);
  int lane  = threadIdx.x & 31;
  const float* xt = x + (size_t)token * D_MODEL;

  float acc[N_EXP];
#pragma unroll
  for (int e = 0; e < N_EXP; ++e) acc[e] = 0.0f;
  for (int d = lane; d < D_MODEL; d += 32) {
    float xv = xt[d];
#pragma unroll
    for (int e = 0; e < N_EXP; ++e) acc[e] += xv * Wr[e * D_MODEL + d];
  }
#pragma unroll
  for (int e = 0; e < N_EXP; ++e) {
    float v = acc[e];
#pragma unroll
    for (int off = 16; off > 0; off >>= 1) v += __shfl_xor(v, off, 32);
    acc[e] = v;
  }
  if (lane == 0) {
    float logits[N_EXP];
    int best = 0;
#pragma unroll
    for (int e = 0; e < N_EXP; ++e) logits[e] = acc[e] + br[e];
    float mx = logits[0];
#pragma unroll
    for (int e = 1; e < N_EXP; ++e)
      if (logits[e] > mx) { mx = logits[e]; best = e; }
    float s = 0.0f;
#pragma unroll
    for (int e = 0; e < N_EXP; ++e) s += __expf(logits[e] - mx);
    idx[token]  = best;
    topp[token] = 1.0f / s;
  }
}

// ---------------------------------------------------------------------------
// Kernel 2: init gather list with -1 padding sentinel
// ---------------------------------------------------------------------------
__global__ void init_perm_kernel(int* __restrict__ perm, int n) {
  int i = blockIdx.x * 256 + threadIdx.x;
  if (i < n) perm[i] = -1;
}

// ---------------------------------------------------------------------------
// Kernel 3: deterministic count/sum reduction -> 32-padded offsets, row count,
//           aux loss (written straight to d_out tail).
// ---------------------------------------------------------------------------
__global__ __launch_bounds__(256) void scan_kernel(
    const int* __restrict__ idx, const float* __restrict__ topp,
    int* __restrict__ offsets, int* __restrict__ cursors,
    int* __restrict__ numRows, float* __restrict__ loss_out) {
  __shared__ int   scnt[256 * N_EXP];
  __shared__ float ssum[256 * N_EXP];
  int tid = threadIdx.x;

  int   cnt[N_EXP];
  float sm[N_EXP];
#pragma unroll
  for (int e = 0; e < N_EXP; ++e) { cnt[e] = 0; sm[e] = 0.0f; }
  for (int t = tid; t < T_TOK; t += 256) {
    int e = idx[t];
    cnt[e] += 1;
    sm[e]  += topp[t];
  }
#pragma unroll
  for (int e = 0; e < N_EXP; ++e) {
    scnt[tid * N_EXP + e] = cnt[e];
    ssum[tid * N_EXP + e] = sm[e];
  }
  __syncthreads();
  for (int s = 128; s > 0; s >>= 1) {
    if (tid < s) {
#pragma unroll
      for (int e = 0; e < N_EXP; ++e) {
        scnt[tid * N_EXP + e] += scnt[(tid + s) * N_EXP + e];
        ssum[tid * N_EXP + e] += ssum[(tid + s) * N_EXP + e];
      }
    }
    __syncthreads();
  }
  if (tid == 0) {
    const float Tf = (float)T_TOK;
    int   off  = 0;
    float loss = 0.0f;
    offsets[0] = 0;
#pragma unroll
    for (int e = 0; e < N_EXP; ++e) {
      int   c = scnt[e];
      float s = ssum[e];
      loss += ((float)c / Tf) * (s / (Tf * Tf));
      off  += ((c + 31) >> 5) << 5;          // pad each expert to 32 rows
      offsets[e + 1] = off;
      cursors[e] = 0;
    }
    *numRows  = off;
    *loss_out = loss * LOSS_SCALE * (float)N_EXP;
  }
}

// ---------------------------------------------------------------------------
// Kernel 4: scatter token ids into expert slots (output order-invariant)
// ---------------------------------------------------------------------------
__global__ void scatter_kernel(const int* __restrict__ idx,
                               const int* __restrict__ offsets,
                               int* __restrict__ cursors,
                               int* __restrict__ perm) {
  int t = blockIdx.x * 256 + threadIdx.x;
  int e = idx[t];
  int pos = atomicAdd(&cursors[e], 1);
  perm[offsets[e] + pos] = t;
}

// ---------------------------------------------------------------------------
// Kernel 5: one-time fp32 -> bf16 hi/lo split (memory-bound pre-pass)
// ---------------------------------------------------------------------------
__global__ __launch_bounds__(256) void split_kernel(
    const float* __restrict__ src, __bf16* __restrict__ hi,
    __bf16* __restrict__ lo, int n4) {
  int i = blockIdx.x * 256 + threadIdx.x;
  if (i >= n4) return;
  float4 v = ((const float4*)src)[i];
  float vv[4] = {v.x, v.y, v.z, v.w};
  v4bf h, l;
#pragma unroll
  for (int j = 0; j < 4; ++j) {
    __bf16 hh = (__bf16)vv[j];
    h[j] = hh;
    l[j] = (__bf16)(vv[j] - (float)hh);
  }
  ((v4bf*)hi)[i] = h;
  ((v4bf*)lo)[i] = l;
}

// ---------------------------------------------------------------------------
// Kernel 6 (main, split path): grouped GEMM on pre-split bf16 hi/lo.
// Block = 256 thr = 8 waves in 2x4 grid; tile = 32 tokens x 256 cols.
// Each wave: 4 accumulators, 12 v_wmma_f32_16x16x32_bf16 per K-step.
// Fragments come from LDS as 2x ds_load_b128 per matrix half (documented
// CDNA5 wave32 layouts: A split-K halves, B per-lane-group K halves).
// ---------------------------------------------------------------------------
__global__ __launch_bounds__(256) void moe_gemm_split_kernel(
    const __bf16* __restrict__ xhi, const __bf16* __restrict__ xlo,
    const __bf16* __restrict__ whi, const __bf16* __restrict__ wlo,
    const float* __restrict__ b, const int* __restrict__ perm,
    const int* __restrict__ offsets, const int* __restrict__ numRows,
    const float* __restrict__ topp, float* __restrict__ y) {
  __shared__ __align__(16) __bf16 sA[2][BM][LDB];   //  5 KB (hi/lo planes)
  __shared__ __align__(16) __bf16 sB[2][BN][LDB];   // 40 KB

  int rowStart = blockIdx.y * BM;
  if (rowStart >= *numRows) return;                 // block-uniform exit

  int e = 0;
#pragma unroll
  for (int i = 0; i < N_EXP - 1; ++i)
    if (rowStart >= offsets[i + 1]) e = i + 1;

  int tid  = threadIdx.x, lane = tid & 31, wave = tid >> 5;
  int mw   = wave >> 2, nw = wave & 3;              // 2x4 wave grid
  int nBase = blockIdx.x * BN;
  int mrow = lane & 15, half = lane >> 4;

  // --- staging assignment: thread -> (row, hi/lo plane, 16B chunk)
  int ur = tid >> 3, rem = tid & 7;
  int plane = rem >> 2, chunk = rem & 3;

  int tokA = perm[rowStart + ur];                   // A row for staging
  const __bf16* aPtr =
      (plane ? xlo : xhi) + (size_t)(tokA >= 0 ? tokA : 0) * D_MODEL + chunk * 8;
  const __bf16* bPtr =
      (plane ? wlo : whi) + (size_t)e * D_MODEL * D_MODEL +
      (size_t)(nBase + ur) * D_MODEL + chunk * 8;

  union Frag { v16bf v; uint4 q[2]; };
  v8f acc[4];
#pragma unroll
  for (int t = 0; t < 4; ++t)
    acc[t] = (v8f){0.f, 0.f, 0.f, 0.f, 0.f, 0.f, 0.f, 0.f};

  for (int k0 = 0; k0 < D_MODEL; k0 += 32) {
    // stage A: 32 rows x 64B x 2 planes = 256 x 16B units (1 per thread)
    uint4 aval = *(const uint4*)(aPtr + k0);
    if (tokA < 0) aval = (uint4){0u, 0u, 0u, 0u};
    *(uint4*)&sA[plane][ur][chunk * 8] = aval;
    // stage B: 256 rows x 64B x 2 planes = 2048 x 16B units (8 per thread)
#pragma unroll
    for (int i = 0; i < 8; ++i) {
      uint4 bval = *(const uint4*)(bPtr + (size_t)i * 32 * D_MODEL + k0);
      *(uint4*)&sB[plane][ur + 32 * i][chunk * 8] = bval;
    }
    __syncthreads();

    // A fragments: lanes<16 -> K {0-7,16-23}, lanes>=16 -> K {8-15,24-31}
    Frag aHi, aLo;
    {
      const char* r0 = (const char*)&sA[0][mw * 16 + mrow][0];
      const char* r1 = (const char*)&sA[1][mw * 16 + mrow][0];
      aHi.q[0] = *(const uint4*)(r0 + 16 * half);
      aHi.q[1] = *(const uint4*)(r0 + 32 + 16 * half);
      aLo.q[0] = *(const uint4*)(r1 + 16 * half);
      aLo.q[1] = *(const uint4*)(r1 + 32 + 16 * half);
    }
#pragma unroll
    for (int t = 0; t < 4; ++t) {
      // B fragments: lanes<16 -> K {0-7,16-23}, lanes>=16 -> K {8-15,24-31}
      Frag bHi, bLo;
      const char* r0 = (const char*)&sB[0][nw * 64 + t * 16 + mrow][0];
      const char* r1 = (const char*)&sB[1][nw * 64 + t * 16 + mrow][0];
      bHi.q[0] = *(const uint4*)(r0 + 16 * half);
      bHi.q[1] = *(const uint4*)(r0 + 32 + 16 * half);
      bLo.q[0] = *(const uint4*)(r1 + 16 * half);
      bLo.q[1] = *(const uint4*)(r1 + 32 + 16 * half);
      // split precision: hi*hi + hi*lo + lo*hi
      acc[t] = __builtin_amdgcn_wmma_f32_16x16x32_bf16(
          false, aHi.v, false, bHi.v, (short)0, acc[t], false, false);
      acc[t] = __builtin_amdgcn_wmma_f32_16x16x32_bf16(
          false, aHi.v, false, bLo.v, (short)0, acc[t], false, false);
      acc[t] = __builtin_amdgcn_wmma_f32_16x16x32_bf16(
          false, aLo.v, false, bHi.v, (short)0, acc[t], false, false);
    }
    __syncthreads();
  }

  // epilogue: C/D layout lane=(n,half), VGPR j -> m = half*8 + j
  int toks[8];
  float gs[8];
#pragma unroll
  for (int j = 0; j < 8; ++j) {
    int tok = perm[rowStart + mw * 16 + half * 8 + j];
    toks[j] = tok;
    gs[j] = (tok >= 0) ? topp[tok] : 0.0f;
  }
#pragma unroll
  for (int t = 0; t < 4; ++t) {
    int nG = nBase + nw * 64 + t * 16 + mrow;
    float bias = b[e * D_MODEL + nG];
#pragma unroll
    for (int j = 0; j < 8; ++j) {
      if (toks[j] >= 0)
        y[(size_t)toks[j] * D_MODEL + nG] = (acc[t][j] + bias) * gs[j];
    }
  }
}

// ---------------------------------------------------------------------------
// Kernel 6b (fallback if workspace too small): on-the-fly bf16x2 split GEMM,
// 16-token x 128-col tiles (round-1 design, 32-padded groups still valid).
// ---------------------------------------------------------------------------
__global__ __launch_bounds__(256) void moe_gemm_fly_kernel(
    const float* __restrict__ x, const float* __restrict__ W,
    const float* __restrict__ b, const int* __restrict__ perm,
    const int* __restrict__ offsets, const int* __restrict__ numRows,
    const float* __restrict__ topp, float* __restrict__ y) {
  __shared__ float lA[16 * 33];
  __shared__ float lB[128 * 33];

  int rowStart = blockIdx.y << 4;
  if (rowStart >= *numRows) return;

  int e = 0;
#pragma unroll
  for (int i = 0; i < N_EXP - 1; ++i)
    if (rowStart >= offsets[i + 1]) e = i + 1;

  int tid = threadIdx.x, lane = tid & 31, wave = tid >> 5;
  int nBase = blockIdx.x * 128;
  int mn = lane & 15, kbA = (lane >> 4) * 8, kbB = (lane >> 4) * 16;
  const float* Wexp = W + (size_t)e * D_MODEL * D_MODEL;

  v8f acc = {0.f, 0.f, 0.f, 0.f, 0.f, 0.f, 0.f, 0.f};
  for (int k0 = 0; k0 < D_MODEL; k0 += 32) {
#pragma unroll
    for (int i = 0; i < 2; ++i) {
      int l = tid + i * 256, m = l >> 5, kk = l & 31;
      int tok = perm[rowStart + m];
      lA[m * 33 + kk] = (tok >= 0) ? x[(size_t)tok * D_MODEL + k0 + kk] : 0.0f;
    }
#pragma unroll
    for (int i = 0; i < 16; ++i) {
      int l = tid + i * 256, n = l >> 5, kk = l & 31;
      lB[n * 33 + kk] = Wexp[(size_t)(nBase + n) * D_MODEL + k0 + kk];
    }
    __syncthreads();
    v16bf a_hi, a_lo, b_hi, b_lo;
#pragma unroll
    for (int j = 0; j < 8; ++j) {
#pragma unroll
      for (int h = 0; h < 2; ++h) {
        int kA = ((j & 4) ? 16 : 0) + kbA + 2 * (j & 3) + h;
        float av = lA[mn * 33 + kA];
        __bf16 ah = (__bf16)av;
        a_hi[2 * j + h] = ah;
        a_lo[2 * j + h] = (__bf16)(av - (float)ah);
        int kB = kA;
        float bv = lB[(wave * 16 + mn) * 33 + kB];
        __bf16 bh = (__bf16)bv;
        b_hi[2 * j + h] = bh;
        b_lo[2 * j + h] = (__bf16)(bv - (float)bh);
      }
    }
    acc = __builtin_amdgcn_wmma_f32_16x16x32_bf16(false, a_hi, false, b_hi,
                                                  (short)0, acc, false, false);
    acc = __builtin_amdgcn_wmma_f32_16x16x32_bf16(false, a_hi, false, b_lo,
                                                  (short)0, acc, false, false);
    acc = __builtin_amdgcn_wmma_f32_16x16x32_bf16(false, a_lo, false, b_hi,
                                                  (short)0, acc, false, false);
    __syncthreads();
  }
  int nG = nBase + wave * 16 + mn;
  float bias = b[e * D_MODEL + nG];
#pragma unroll
  for (int j = 0; j < 8; ++j) {
    int tok = perm[rowStart + (lane >> 4) * 8 + j];
    if (tok >= 0)
      y[(size_t)tok * D_MODEL + nG] = (acc[j] + bias) * topp[tok];
  }
}

// ---------------------------------------------------------------------------
extern "C" void kernel_launch(void* const* d_in, const int* in_sizes, int n_in,
                              void* d_out, int out_size, void* d_ws, size_t ws_size,
                              hipStream_t stream) {
  const float* x  = (const float*)d_in[0];
  const float* Wr = (const float*)d_in[1];
  const float* br = (const float*)d_in[2];
  const float* W  = (const float*)d_in[3];
  const float* b  = (const float*)d_in[4];

  float* y    = (float*)d_out;
  float* loss = y + (size_t)T_TOK * D_MODEL;   // tuple output: y, then loss

  // workspace: metadata (<128KB), then optional bf16 hi/lo mirrors (~192MB)
  int* ws_i     = (int*)d_ws;
  int* idx      = ws_i;
  int* offsets  = ws_i + T_TOK;
  int* cursors  = offsets + (N_EXP + 1);
  int* numRows  = cursors + N_EXP;
  int* perm     = numRows + 1;
  float* topp   = (float*)(perm + PERM_CAP);

  const size_t META = 128 * 1024;
  const size_t XN = (size_t)T_TOK * D_MODEL;         // 16,777,216
  const size_t WN = (size_t)N_EXP * D_MODEL * D_MODEL;
  __bf16* xhi = (__bf16*)((char*)d_ws + META);
  __bf16* xlo = xhi + XN;
  __bf16* whi = xlo + XN;
  __bf16* wlo = whi + WN;
  const size_t NEED = META + 2 * (XN + WN) * sizeof(__bf16);
  const bool use_split = ws_size >= NEED;

  router_kernel<<<T_TOK / 8, 256, 0, stream>>>(x, Wr, br, idx, topp);
  init_perm_kernel<<<(PERM_CAP + 255) / 256, 256, 0, stream>>>(perm, PERM_CAP);
  scan_kernel<<<1, 256, 0, stream>>>(idx, topp, offsets, cursors, numRows, loss);
  scatter_kernel<<<T_TOK / 256, 256, 0, stream>>>(idx, offsets, cursors, perm);

  if (use_split) {
    split_kernel<<<(int)(XN / 4 / 256), 256, 0, stream>>>(x, xhi, xlo, (int)(XN / 4));
    split_kernel<<<(int)(WN / 4 / 256), 256, 0, stream>>>(W, whi, wlo, (int)(WN / 4));
    dim3 grid(D_MODEL / BN, NT32);
    moe_gemm_split_kernel<<<grid, 256, 0, stream>>>(xhi, xlo, whi, wlo, b, perm,
                                                    offsets, numRows, topp, y);
  } else {
    dim3 grid(D_MODEL / 128, NT16);
    moe_gemm_fly_kernel<<<grid, 256, 0, stream>>>(x, W, b, perm, offsets,
                                                  numRows, topp, y);
  }
}
